// PixelEachSubstitutor_64811056496816
// MI455X (gfx1250) — matrix operand order repaired and checked by reading, not compiled, not run on hardware
//
#include <hip/hip_runtime.h>

typedef __attribute__((ext_vector_type(16))) _Float16 v16h;
typedef __attribute__((ext_vector_type(8)))  float    v8f;

#define NB 16
#define CC 10    // classes
#define HH 30
#define WW 30
#define DD 11    // d = C+1
#define LL 81    // MAX_V (sequence length)
#define LP 96    // padded L (6 x 16)
#define VV 32

// ---------------- WMMA helpers (CDNA5 gfx1250, wave32) ----------------

static __device__ __forceinline__ v8f wmma_f16(v16h a, v16h b, v8f c) {
  // (neg_a, A, neg_b, B, c_mod, C, reuse_a, reuse_b)
  return __builtin_amdgcn_wmma_f32_16x16x32_f16(false, a, false, b, (short)0, c,
                                                false, false);
}

// A fragment 16x32 (MxK) from row-major f32, zero-fill guards.
static __device__ __forceinline__ v16h frag_A_f32(const float* S, int ld,
                                                  int m0, int k0, int mMax, int kMax) {
  const int lane = threadIdx.x & 31;
  const int m = m0 + (lane & 15);
  const int kb = (lane >> 4) << 3;              // 0 or 8
  v16h a;
#pragma unroll
  for (int r = 0; r < 8; ++r) {
    const int kk = ((r & 4) << 2) + kb + ((r & 3) << 1);  // 16*(r>>2) + kb + 2*(r&3)
#pragma unroll
    for (int h = 0; h < 2; ++h) {
      const int k = k0 + kk + h;
      const float v = (m < mMax && k < kMax) ? S[m * ld + k] : 0.0f;
      a[2 * r + h] = (_Float16)v;
    }
  }
  return a;
}

// A fragment with transposed read: element (m,k) = S[k*ld + m]
static __device__ __forceinline__ v16h frag_AT_f32(const float* S, int ld,
                                                   int m0, int k0, int mMax, int kMax) {
  const int lane = threadIdx.x & 31;
  const int m = m0 + (lane & 15);
  const int kb = (lane >> 4) << 3;
  v16h a;
#pragma unroll
  for (int r = 0; r < 8; ++r) {
    const int kk = ((r & 4) << 2) + kb + ((r & 3) << 1);
#pragma unroll
    for (int h = 0; h < 2; ++h) {
      const int k = k0 + kk + h;
      const float v = (m < mMax && k < kMax) ? S[k * ld + m] : 0.0f;
      a[2 * r + h] = (_Float16)v;
    }
  }
  return a;
}

// A fragment from f16 storage, no guards (fully initialized padded matrix)
static __device__ __forceinline__ v16h frag_A_f16(const _Float16* S, int ld,
                                                  int m0, int k0) {
  const int lane = threadIdx.x & 31;
  const int m = m0 + (lane & 15);
  const int kb = (lane >> 4) << 3;
  v16h a;
#pragma unroll
  for (int r = 0; r < 8; ++r) {
    const int kk = ((r & 4) << 2) + kb + ((r & 3) << 1);
    a[2 * r]     = S[m * ld + k0 + kk];
    a[2 * r + 1] = S[m * ld + k0 + kk + 1];
  }
  return a;
}

// A fragment from f16 storage with K guard (ld < 32 padded buffers)
static __device__ __forceinline__ v16h frag_A_f16g(const _Float16* S, int ld,
                                                   int m0, int kMax) {
  const int lane = threadIdx.x & 31;
  const int m = m0 + (lane & 15);
  const int kb = (lane >> 4) << 3;
  v16h a;
#pragma unroll
  for (int r = 0; r < 8; ++r) {
    const int kk = ((r & 4) << 2) + kb + ((r & 3) << 1);
#pragma unroll
    for (int h = 0; h < 2; ++h) {
      const int k = kk + h;
      a[2 * r + h] = (k < kMax) ? S[m * ld + k] : (_Float16)0.f;
    }
  }
  return a;
}

// B fragment 32x16 (KxN) with B(k,n) = W[n*ldw + k]  (i.e. Y = X @ W^T), f32 src
static __device__ __forceinline__ v16h frag_B_WT(const float* W, int ldw,
                                                 int n0, int k0, int nMax, int kMax) {
  const int lane = threadIdx.x & 31;
  const int n = n0 + (lane & 15);
  const int kb = (lane >> 4) << 3;              // 0 or 8
  v16h b;
#pragma unroll
  for (int r = 0; r < 8; ++r) {
#pragma unroll
    for (int h = 0; h < 2; ++h) {
      const int k = k0 + kb + ((r & 4) << 2) + ((r & 3) << 1) + h;
      const float v = (n < nMax && k < kMax) ? W[n * ldw + k] : 0.0f;
      b[2 * r + h] = (_Float16)v;
    }
  }
  return b;
}

// B fragment with B(k,n) = S[n*ld + k], f16 source, K guard
static __device__ __forceinline__ v16h frag_B_WT_h(const _Float16* S, int ld,
                                                   int n0, int kMax) {
  const int lane = threadIdx.x & 31;
  const int n = n0 + (lane & 15);
  const int kb = (lane >> 4) << 3;
  v16h b;
#pragma unroll
  for (int r = 0; r < 8; ++r) {
#pragma unroll
    for (int h = 0; h < 2; ++h) {
      const int k = kb + ((r & 4) << 2) + ((r & 3) << 1) + h;
      b[2 * r + h] = (k < kMax) ? S[n * ld + k] : (_Float16)0.f;
    }
  }
  return b;
}

// B fragment with B(k,n) = S[k*ld + n], f16 source, no guards
static __device__ __forceinline__ v16h frag_B_RM_h(const _Float16* S, int ld,
                                                   int k0, int n0) {
  const int lane = threadIdx.x & 31;
  const int n = n0 + (lane & 15);
  const int kb = (lane >> 4) << 3;
  v16h b;
#pragma unroll
  for (int r = 0; r < 8; ++r) {
#pragma unroll
    for (int h = 0; h < 2; ++h) {
      const int k = k0 + kb + ((r & 4) << 2) + ((r & 3) << 1) + h;
      b[2 * r + h] = S[k * ld + n];
    }
  }
  return b;
}

static __device__ __forceinline__ void store_C(float* D, int ld, int m0, int n0,
                                               v8f c, float scale) {
  const int lane = threadIdx.x & 31;
  const int nn = n0 + (lane & 15);
  const int mb = m0 + ((lane >> 4) << 3);
#pragma unroll
  for (int r = 0; r < 8; ++r) D[(mb + r) * ld + nn] = c[r] * scale;
}

static __device__ __forceinline__ void store_Ch(_Float16* D, int ld, int m0, int n0,
                                                v8f c, float scale) {
  const int lane = threadIdx.x & 31;
  const int nn = n0 + (lane & 15);
  const int mb = m0 + ((lane >> 4) << 3);
#pragma unroll
  for (int r = 0; r < 8; ++r) D[(mb + r) * ld + nn] = (_Float16)(c[r] * scale);
}

// LayerNorm of Y += Add over `rows` rows of length `cols`, weight w, eps 1e-5
static __device__ __forceinline__ void ln_add_rows(float* Y, const float* Add,
                                                   const float* w, int rows, int cols) {
  const int lane = threadIdx.x & 31;
  for (int i = lane; i < rows; i += 32) {
    float mean = 0.f;
    for (int c = 0; c < cols; ++c) {
      const float v = Y[i * cols + c] + Add[i * cols + c];
      Y[i * cols + c] = v;
      mean += v;
    }
    mean /= (float)cols;
    float var = 0.f;
    for (int c = 0; c < cols; ++c) {
      const float d = Y[i * cols + c] - mean;
      var += d * d;
    }
    var /= (float)cols;
    const float rs = rsqrtf(var + 1e-5f);
    for (int c = 0; c < cols; ++c)
      Y[i * cols + c] = (Y[i * cols + c] - mean) * rs * w[c];
  }
}

// ---------------- kernel: one wave32 per pixel ----------------
// LDS: F 6144B (fp32 master for residual/LN) + Qh/Kh/Vh 3*3072B (f16 WMMA
// operands) + Sh 18432B (f16 scores, reused for out-proj scratch + decoder)
// = 33792B per workgroup  ->  ~9 workgroups resident per 320KB WGP.

__global__ __launch_bounds__(32) void pixel_transformer_kernel(
    const float* __restrict__ x, const float* __restrict__ V_feature,
    const float* __restrict__ enc_in_proj, const float* __restrict__ enc_out_proj,
    const float* __restrict__ enc_lin1, const float* __restrict__ enc_lin2,
    const float* __restrict__ enc_ln1, const float* __restrict__ enc_ln2,
    const float* __restrict__ ffV_w,
    const float* __restrict__ dec_sa_in, const float* __restrict__ dec_sa_out,
    const float* __restrict__ dec_ca_in, const float* __restrict__ dec_ca_out,
    const float* __restrict__ dec_lin1, const float* __restrict__ dec_lin2,
    const float* __restrict__ dec_ln1, const float* __restrict__ dec_ln2,
    const float* __restrict__ dec_ln3,
    const float* __restrict__ W_d0, const float* __restrict__ W_d1,
    float* __restrict__ out) {
  __shared__ float F[LP * 16];                      // tokens f (96x16 fp32)
  __shared__ __align__(16) _Float16 Qh[LP * 16];    // Q (f16); later attn-out O
  __shared__ __align__(16) _Float16 Kh[LP * 16];    // K (f16)
  __shared__ __align__(16) _Float16 Vh[LP * 16];    // V (f16)
  __shared__ __align__(16) _Float16 Sh[LP * LP];    // 96x96 scores (f16); reused

  float* DEC  = (float*)Sh;        // region reuse once scores are dead
  float* memb = DEC;               // mem: 16x32 fp32
  float* yb   = DEC + 512;         // y:   16x32 fp32
  float* tq   = DEC + 1024;
  float* tk   = DEC + 1536;
  float* tv   = DEC + 2048;
  float* to   = DEC + 2560;        // DEC[0..3071] used by decoder
  float* Pb   = DEC + 1024;        // out-proj result 96x16 fp32 (dead before tq)

  const int lane = threadIdx.x & 31;
  const int p  = blockIdx.x;
  const int n  = p / (HH * WW);
  const int ij = p % (HH * WW);
  const int pi = ij / WW;
  const int pj = ij % WW;

  __builtin_prefetch(&x[((size_t)n * CC * HH + pi) * WW + pj], 0, 1);

  // ---- build f (81x11) padded to 96x16; rows 9..95 and cols 11..15 zero ----
  for (int idx = lane; idx < LP * 16; idx += 32) F[idx] = 0.0f;
  __syncthreads();
  for (int idx = lane; idx < 9 * DD; idx += 32) {
    const int t = idx / DD, c = idx % DD;
    float v;
    if (c == CC) {
      v = 1.0f;                                  // ones channel (un-padded)
    } else {
      const int ii = pi + t / 3 - 1, jj = pj + t % 3 - 1;
      v = (ii >= 0 && ii < HH && jj >= 0 && jj < WW)
              ? x[((n * CC + c) * HH + ii) * WW + jj]
              : 0.0f;
    }
    F[t * 16 + c] = v;
  }
  __syncthreads();

  // ---- encoder QKV projections (f @ W^T), K-pad 11->32, f16 results ----
  {
    const v16h bq = frag_B_WT(enc_in_proj, DD, 0, 0, DD, DD);
    const v16h bk = frag_B_WT(enc_in_proj + DD * DD, DD, 0, 0, DD, DD);
    const v16h bv = frag_B_WT(enc_in_proj + 2 * DD * DD, DD, 0, 0, DD, DD);
    for (int mt = 0; mt < 6; ++mt) {
      const v16h a = frag_A_f32(F, 16, mt * 16, 0, LP, 16);
      v8f z = {};
      store_Ch(Qh, 16, mt * 16, 0, wmma_f16(a, bq, z), 1.0f);
      store_Ch(Kh, 16, mt * 16, 0, wmma_f16(a, bk, z), 1.0f);
      store_Ch(Vh, 16, mt * 16, 0, wmma_f16(a, bv, z), 1.0f);
    }
  }
  __syncthreads();

  // ---- scores = Q K^T / sqrt(11) -> Sh (f16) ----
  {
    const float sc = 0.30151134f;  // 1/sqrt(11)
    for (int mt = 0; mt < 6; ++mt) {
      const v16h a = frag_A_f16g(Qh, 16, mt * 16, 16);
      for (int nt = 0; nt < 6; ++nt) {
        const v16h b = frag_B_WT_h(Kh, 16, nt * 16, 16);
        v8f z = {};
        store_Ch(Sh, LP, mt * 16, nt * 16, wmma_f16(a, b, z), sc);
      }
    }
  }
  __syncthreads();

  // ---- softmax over j<81 per row; zero padding rows/cols ----
  for (int i = lane; i < LP; i += 32) {
    if (i < LL) {
      float mx = -1e30f;
      for (int j = 0; j < LL; ++j) mx = fmaxf(mx, (float)Sh[i * LP + j]);
      float sum = 0.f;
      for (int j = 0; j < LL; ++j) sum += __expf((float)Sh[i * LP + j] - mx);
      const float inv = 1.0f / sum;
      for (int j = 0; j < LL; ++j)
        Sh[i * LP + j] = (_Float16)(__expf((float)Sh[i * LP + j] - mx) * inv);
      for (int j = LL; j < LP; ++j) Sh[i * LP + j] = (_Float16)0.f;
    } else {
      for (int j = 0; j < LP; ++j) Sh[i * LP + j] = (_Float16)0.f;
    }
  }
  __syncthreads();

  // ---- O = attn @ V  (K = 96 in 3 chunks) -> Qh (f16) ----
  {
    const v16h b0 = frag_B_RM_h(Vh, 16, 0, 0);
    const v16h b1 = frag_B_RM_h(Vh, 16, 32, 0);
    const v16h b2 = frag_B_RM_h(Vh, 16, 64, 0);
    for (int mt = 0; mt < 6; ++mt) {
      v8f c = {};
      c = wmma_f16(frag_A_f16(Sh, LP, mt * 16, 0), b0, c);
      c = wmma_f16(frag_A_f16(Sh, LP, mt * 16, 32), b1, c);
      c = wmma_f16(frag_A_f16(Sh, LP, mt * 16, 64), b2, c);
      store_Ch(Qh, 16, mt * 16, 0, c, 1.0f);
    }
  }
  __syncthreads();

  // ---- out_proj -> Pb (fp32, in dead score region) ----
  {
    const v16h b = frag_B_WT(enc_out_proj, DD, 0, 0, DD, DD);
    for (int mt = 0; mt < 6; ++mt) {
      const v16h a = frag_A_f16g(Qh, 16, mt * 16, 16);
      v8f z = {};
      store_C(Pb, 16, mt * 16, 0, wmma_f16(a, b, z), 1.0f);
    }
  }
  __syncthreads();

  // ---- residual + LN1, then FF + LN2 per row (fp32) ----
  for (int i = lane; i < LL; i += 32) {
    float v[DD], w2[DD];
    float mean = 0.f;
#pragma unroll
    for (int c = 0; c < DD; ++c) { v[c] = F[i * 16 + c] + Pb[i * 16 + c]; mean += v[c]; }
    mean /= (float)DD;
    float var = 0.f;
#pragma unroll
    for (int c = 0; c < DD; ++c) { const float d = v[c] - mean; var += d * d; }
    var /= (float)DD;
    const float rs = rsqrtf(var + 1e-5f);
#pragma unroll
    for (int c = 0; c < DD; ++c) v[c] = (v[c] - mean) * rs * enc_ln1[c];
    float h = 0.f;
#pragma unroll
    for (int c = 0; c < DD; ++c) h += v[c] * enc_lin1[c];
    h = fmaxf(h, 0.f);
    mean = 0.f;
#pragma unroll
    for (int c = 0; c < DD; ++c) { w2[c] = v[c] + h * enc_lin2[c]; mean += w2[c]; }
    mean /= (float)DD;
    var = 0.f;
#pragma unroll
    for (int c = 0; c < DD; ++c) { const float d = w2[c] - mean; var += d * d; }
    var /= (float)DD;
    const float r2 = rsqrtf(var + 1e-5f);
#pragma unroll
    for (int c = 0; c < DD; ++c) F[i * 16 + c] = (w2[c] - mean) * r2 * enc_ln2[c];
  }
  __syncthreads();

  // ---- mem = f^T(11x81) @ ffV_w^T(81x32) -> 11x32 (into DEC region) ----
  for (int nt = 0; nt < 2; ++nt) {
    v8f c = {};
    for (int kc = 0; kc < 3; ++kc) {
      const v16h a = frag_AT_f32(F, 16, 0, kc * 32, 16, LP);
      const v16h b = frag_B_WT(ffV_w, LL, nt * 16, kc * 32, VV, LL);
      c = wmma_f16(a, b, c);
    }
    store_C(memb, VV, 0, nt * 16, c, 1.0f);
  }
  __syncthreads();

  // ---- decoder init: y = V_feature (10x32), rows 10..15 zero ----
  for (int idx = lane; idx < 16 * VV; idx += 32)
    yb[idx] = (idx < CC * VV) ? V_feature[idx] : 0.0f;
  __syncthreads();

  // ---- decoder self-attention (32 heads, head-dim 1) ----
  {
    const v16h a = frag_A_f32(yb, VV, 0, 0, 16, VV);
    for (int nt = 0; nt < 2; ++nt) {
      v8f z = {};
      store_C(tq, VV, 0, nt * 16,
              wmma_f16(a, frag_B_WT(dec_sa_in, VV, nt * 16, 0, VV, VV), z), 1.0f);
      store_C(tk, VV, 0, nt * 16,
              wmma_f16(a, frag_B_WT(dec_sa_in + VV * VV, VV, nt * 16, 0, VV, VV), z), 1.0f);
      store_C(tv, VV, 0, nt * 16,
              wmma_f16(a, frag_B_WT(dec_sa_in + 2 * VV * VV, VV, nt * 16, 0, VV, VV), z), 1.0f);
    }
  }
  __syncthreads();
  {
    const int h = lane;  // one head per lane
    for (int i = 0; i < CC; ++i) {
      const float qv = tq[i * VV + h];
      float s[CC];
      float mx = -1e30f;
#pragma unroll
      for (int j = 0; j < CC; ++j) { s[j] = qv * tk[j * VV + h]; mx = fmaxf(mx, s[j]); }
      float sum = 0.f;
#pragma unroll
      for (int j = 0; j < CC; ++j) { s[j] = __expf(s[j] - mx); sum += s[j]; }
      float o = 0.f;
#pragma unroll
      for (int j = 0; j < CC; ++j) o += s[j] * tv[j * VV + h];
      to[i * VV + h] = o / sum;
    }
  }
  __syncthreads();
  {
    const v16h a = frag_A_f32(to, VV, 0, 0, CC, VV);
    for (int nt = 0; nt < 2; ++nt) {
      v8f z = {};
      store_C(tq, VV, 0, nt * 16,
              wmma_f16(a, frag_B_WT(dec_sa_out, VV, nt * 16, 0, VV, VV), z), 1.0f);
    }
  }
  __syncthreads();
  ln_add_rows(yb, tq, dec_ln1, CC, VV);
  __syncthreads();

  // ---- decoder cross-attention (q from y, k/v from mem, Lk=11) ----
  {
    const v16h ay = frag_A_f32(yb, VV, 0, 0, 16, VV);
    const v16h am = frag_A_f32(memb, VV, 0, 0, 16, VV);
    for (int nt = 0; nt < 2; ++nt) {
      v8f z = {};
      store_C(tq, VV, 0, nt * 16,
              wmma_f16(ay, frag_B_WT(dec_ca_in, VV, nt * 16, 0, VV, VV), z), 1.0f);
      store_C(tk, VV, 0, nt * 16,
              wmma_f16(am, frag_B_WT(dec_ca_in + VV * VV, VV, nt * 16, 0, VV, VV), z), 1.0f);
      store_C(tv, VV, 0, nt * 16,
              wmma_f16(am, frag_B_WT(dec_ca_in + 2 * VV * VV, VV, nt * 16, 0, VV, VV), z), 1.0f);
    }
  }
  __syncthreads();
  {
    const int h = lane;
    for (int i = 0; i < CC; ++i) {
      const float qv = tq[i * VV + h];
      float s[DD];
      float mx = -1e30f;
#pragma unroll
      for (int j = 0; j < DD; ++j) { s[j] = qv * tk[j * VV + h]; mx = fmaxf(mx, s[j]); }
      float sum = 0.f;
#pragma unroll
      for (int j = 0; j < DD; ++j) { s[j] = __expf(s[j] - mx); sum += s[j]; }
      float o = 0.f;
#pragma unroll
      for (int j = 0; j < DD; ++j) o += s[j] * tv[j * VV + h];
      to[i * VV + h] = o / sum;
    }
  }
  __syncthreads();
  {
    const v16h a = frag_A_f32(to, VV, 0, 0, CC, VV);
    for (int nt = 0; nt < 2; ++nt) {
      v8f z = {};
      store_C(tq, VV, 0, nt * 16,
              wmma_f16(a, frag_B_WT(dec_ca_out, VV, nt * 16, 0, VV, VV), z), 1.0f);
    }
  }
  __syncthreads();
  ln_add_rows(yb, tq, dec_ln2, CC, VV);
  __syncthreads();

  // ---- decoder FF + LN3 ----
  for (int i = lane; i < CC; i += 32) {
    float h = 0.f;
#pragma unroll
    for (int c = 0; c < VV; ++c) h += yb[i * VV + c] * dec_lin1[c];
    h = fmaxf(h, 0.f);
#pragma unroll
    for (int c = 0; c < VV; ++c) tq[i * VV + c] = h * dec_lin2[c];
  }
  __syncthreads();
  ln_add_rows(yb, tq, dec_ln3, CC, VV);
  __syncthreads();

  // ---- head: relu(y @ W_d0^T) @ W_d1^T ----
  {
    const v16h a = frag_A_f32(yb, VV, 0, 0, CC, VV);
    const v16h b = frag_B_WT(W_d0, VV, 0, 0, 8, VV);
    v8f z = {};
    store_C(to, VV, 0, 0, wmma_f16(a, b, z), 1.0f);
  }
  __syncthreads();
  if (lane < CC) {
    float o = 0.f;
#pragma unroll
    for (int k = 0; k < 8; ++k) o += fmaxf(to[lane * VV + k], 0.f) * W_d1[k];
    out[((n * CC + lane) * HH + pi) * WW + pj] = o;
  }
}

extern "C" void kernel_launch(void* const* d_in, const int* in_sizes, int n_in,
                              void* d_out, int out_size, void* d_ws, size_t ws_size,
                              hipStream_t stream) {
  (void)in_sizes; (void)n_in; (void)out_size; (void)d_ws; (void)ws_size;
  const float* x           = (const float*)d_in[0];
  const float* V_feature   = (const float*)d_in[1];
  const float* enc_in_proj = (const float*)d_in[2];
  const float* enc_out_proj= (const float*)d_in[3];
  const float* enc_lin1    = (const float*)d_in[4];
  const float* enc_lin2    = (const float*)d_in[5];
  const float* enc_ln1     = (const float*)d_in[6];
  const float* enc_ln2     = (const float*)d_in[7];
  const float* ffV_w       = (const float*)d_in[8];
  const float* dec_sa_in   = (const float*)d_in[9];
  const float* dec_sa_out  = (const float*)d_in[10];
  const float* dec_ca_in   = (const float*)d_in[11];
  const float* dec_ca_out  = (const float*)d_in[12];
  const float* dec_lin1    = (const float*)d_in[13];
  const float* dec_lin2    = (const float*)d_in[14];
  const float* dec_ln1     = (const float*)d_in[15];
  const float* dec_ln2     = (const float*)d_in[16];
  const float* dec_ln3     = (const float*)d_in[17];
  const float* W_d0        = (const float*)d_in[18];
  const float* W_d1        = (const float*)d_in[19];
  float* out = (float*)d_out;

  const int nblocks = NB * HH * WW;  // 14400 pixels, one wave32 each
  pixel_transformer_kernel<<<nblocks, 32, 0, stream>>>(
      x, V_feature, enc_in_proj, enc_out_proj, enc_lin1, enc_lin2, enc_ln1,
      enc_ln2, ffV_w, dec_sa_in, dec_sa_out, dec_ca_in, dec_ca_out, dec_lin1,
      dec_lin2, dec_ln1, dec_ln2, dec_ln3, W_d0, W_d1, out);
}
